// PHGNS2_45672682226298
// MI455X (gfx1250) — matrix operand order repaired and checked by reading, not compiled, not run on hardware
//
#include <hip/hip_runtime.h>
#include <hip/hip_bf16.h>

// ---------------------------------------------------------------------------
// Problem constants
// ---------------------------------------------------------------------------
#define N_NODES 4096
#define N_EDGES 16384
#define LDIM    512
#define MP_STEPS 5
#define DT_STEP 0.01f

typedef __bf16 bf16_t;
typedef __attribute__((ext_vector_type(16))) __bf16 v16bf;
typedef __attribute__((ext_vector_type(8)))  __bf16 v8bf;
typedef __attribute__((ext_vector_type(8)))  float  v8f;

// GEMM flags
#define GF_ACCUM  1   // C += result (read-modify-write)
#define GF_RELU   2   // relu applied after bias/accum
#define GF_ATOMIC 4   // atomicAdd into C[rowmapC[m]]

// Load a 16-element bf16 fragment from two 16-byte-aligned 8-element chunks.
__device__ __forceinline__ v16bf load_frag2(const bf16_t* p0, const bf16_t* p1) {
    p0 = (const bf16_t*)__builtin_assume_aligned(p0, 16);
    p1 = (const bf16_t*)__builtin_assume_aligned(p1, 16);
    v8bf lo = *(const v8bf*)p0;
    v8bf hi = *(const v8bf*)p1;
    return __builtin_shufflevector(lo, hi, 0, 1, 2, 3, 4, 5, 6, 7,
                                   8, 9, 10, 11, 12, 13, 14, 15);
}

// ---------------------------------------------------------------------------
// bf16-WMMA GEMM:
//   acc(m,n) = sum_k A[rowA(m),k] * Bt[n*ldb + k]
//   A: fp32, converted to bf16 while staging to LDS.
//   Bt: bf16, stored [N,K] row-major (weights pre-transposed for forward;
//       plain row-major weights ARE Bt for the backward "@ W^T" GEMMs).
//   epilogue: +bias[n], +=C, relu-mask, relu, atomic row-scatter
// Block = 256 threads (8 wave32) -> 128x64 tile; each wave computes 32x32
// (2 A-frags x 2 B-frags -> 4 WMMA per 32-wide K step). Double-buffered LDS.
// Requires M%128==0, N%64==0, K%32==0 (true for all shapes here).
// ---------------------------------------------------------------------------
__global__ void __launch_bounds__(256)
gemm_bf16_wmma(const float* __restrict__ A, int lda, const int* __restrict__ rowmapA,
               const bf16_t* __restrict__ B, int ldb,
               const float* __restrict__ bias, const float* __restrict__ mask,
               float* __restrict__ C, int ldc, const int* __restrict__ rowmapC,
               int K, int flags)
{
    __shared__ __align__(16) bf16_t sA[2][128][32 + 8];  // [m][k], 80B pitch
    __shared__ __align__(16) bf16_t sB[2][64][32 + 8];   // [n][k], 80B pitch

    const int tid  = threadIdx.x;
    const int lane = tid & 31;
    const int wid  = tid >> 5;
    const int wr   = wid >> 1;   // 0..3 : 32-row slice
    const int wc   = wid & 1;    // 0..1 : 32-col slice
    const int m0   = blockIdx.y * 128;
    const int n0   = blockIdx.x * 64;

    // --- staging assignments (hoisted row pointers; rowmap gather once) ---
    const int ar  = tid >> 1;            // 0..127
    const int ac  = (tid & 1) * 16;      // 0,16
    const int gm  = rowmapA ? rowmapA[m0 + ar] : (m0 + ar);
    const float* aRow = A + (size_t)gm * lda;

    const int bn  = tid >> 2;            // 0..63
    const int bk8 = (tid & 3) * 8;       // 0,8,16,24
    const bf16_t* btRow = B + (size_t)(n0 + bn) * ldb;

    v8f acc[2][2] = {};

    const int nk = K >> 5;

    // stage k-tile `kt` into buffer `buf`
    auto stage = [&](int buf, int kt) {
        const float* pa = aRow + kt * 32 + ac;
        #pragma unroll
        for (int j = 0; j < 16; ++j) sA[buf][ar][ac + j] = (bf16_t)pa[j];
        const bf16_t* pb = btRow + kt * 32 + bk8;
        #pragma unroll
        for (int j = 0; j < 8; ++j) sB[buf][bn][bk8 + j] = pb[j];
    };

    stage(0, 0);
    __syncthreads();

    const int kbA = (lane < 16) ? 0 : 8;    // A frag: K = kbA..kbA+7, 16+kbA..+7
    const int kbB = (lane < 16) ? 0 : 8;    // B frag: K = kbB..kbB+7, 16+kbB..+7

    for (int kt = 0; kt < nk; ++kt) {
        const int cur = kt & 1;
        if (kt + 1 < nk) stage(cur ^ 1, kt + 1);   // overlap staging with WMMA

        v16bf af[2], bfr[2];
        #pragma unroll
        for (int f = 0; f < 2; ++f) {
            const int ml = wr * 32 + f * 16 + (lane & 15);
            af[f] = load_frag2(&sA[cur][ml][kbA], &sA[cur][ml][16 + kbA]);
        }
        #pragma unroll
        for (int g = 0; g < 2; ++g) {
            const int nl = wc * 32 + g * 16 + (lane & 15);
            bfr[g] = load_frag2(&sB[cur][nl][kbB], &sB[cur][nl][16 + kbB]);
        }
        #pragma unroll
        for (int f = 0; f < 2; ++f)
            #pragma unroll
            for (int g = 0; g < 2; ++g)
                acc[f][g] = __builtin_amdgcn_wmma_f32_16x16x32_bf16(
                    false, af[f], false, bfr[g], (short)0, acc[f][g], false, false);

        __syncthreads();
    }

    // --- epilogue: C/D layout: lane -> n = lane%16 ; vgpr r -> m = r + (lane>=16)*8
    #pragma unroll
    for (int f = 0; f < 2; ++f) {
        const int mbase = m0 + wr * 32 + f * 16 + ((lane >= 16) ? 8 : 0);
        #pragma unroll
        for (int g = 0; g < 2; ++g) {
            const int ng = n0 + wc * 32 + g * 16 + (lane & 15);
            #pragma unroll
            for (int r = 0; r < 8; ++r) {
                const int mg = mbase + r;
                float v = acc[f][g][r];
                if (bias) v += bias[ng];
                if (flags & GF_ATOMIC) {
                    atomicAdd(&C[(size_t)rowmapC[mg] * ldc + ng], v);
                } else {
                    const size_t ci = (size_t)mg * ldc + ng;
                    if (flags & GF_ACCUM) v += C[ci];
                    if (mask) v = (mask[ci] > 0.f) ? v : 0.f;
                    if (flags & GF_RELU) v = fmaxf(v, 0.f);
                    C[ci] = v;
                }
            }
        }
    }
}

// ---------------------------------------------------------------------------
// Elementwise / reduction helpers
// ---------------------------------------------------------------------------
__device__ __forceinline__ float wave_sum(float v) {
    #pragma unroll
    for (int off = 16; off > 0; off >>= 1) v += __shfl_xor(v, off, 32);
    return v;
}

__global__ void __launch_bounds__(256)
cvt_f32_to_bf16(const float* __restrict__ src, bf16_t* __restrict__ dst, int n) {
    int i = blockIdx.x * 256 + threadIdx.x;
    if (i < n) dst[i] = (bf16_t)src[i];
}

// transpose-convert: src fp32 [R,C] -> dst bf16 [C,R]
__global__ void __launch_bounds__(256)
cvt_f32_to_bf16_t(const float* __restrict__ src, bf16_t* __restrict__ dst,
                  int R, int Ccols) {
    int i = blockIdx.x * 256 + threadIdx.x;
    if (i >= R * Ccols) return;
    int r = i / Ccols, c = i - r * Ccols;
    dst[(size_t)c * R + r] = (bf16_t)src[i];
}

__global__ void __launch_bounds__(256)
zero_f32(float* __restrict__ p, int n) {
    int i = blockIdx.x * 256 + threadIdx.x;
    if (i < n) p[i] = 0.f;
}

// h1[r,c] = relu(x[r]*W1[c] + b1[c])   (encoder: input dim 1 -> rank-1)
__global__ void __launch_bounds__(256)
enc_rank1_fwd(const float* __restrict__ x, const float* __restrict__ W1,
              const float* __restrict__ b1, float* __restrict__ h1, int rows) {
    size_t i = (size_t)blockIdx.x * 256 + threadIdx.x;
    if (i >= (size_t)rows * LDIM) return;
    int r = (int)(i >> 9), c = (int)(i & (LDIM - 1));
    h1[i] = fmaxf(x[r] * W1[c] + b1[c], 0.f);
}

// Layer norm forward: wave per row; saves y and rstd.
__global__ void __launch_bounds__(256)
ln_fwd(const float* __restrict__ u, float* __restrict__ y,
       float* __restrict__ rstd, int rows) {
    int row = blockIdx.x * 8 + (threadIdx.x >> 5);
    int lane = threadIdx.x & 31;
    if (row >= rows) return;
    const float* x = u + (size_t)row * LDIM;
    float s = 0.f, s2 = 0.f;
    for (int c = lane; c < LDIM; c += 32) { float v = x[c]; s += v; s2 += v * v; }
    s = wave_sum(s); s2 = wave_sum(s2);
    const float mean = s * (1.f / LDIM);
    const float var  = s2 * (1.f / LDIM) - mean * mean;
    const float r    = rsqrtf(var + 1e-6f);
    if (lane == 0) rstd[row] = r;
    float* yo = y + (size_t)row * LDIM;
    for (int c = lane; c < LDIM; c += 32) yo[c] = (x[c] - mean) * r;
}

// LN backward (no affine): dx = rstd*(g - mean(g) - y*mean(g*y))
__global__ void __launch_bounds__(256)
ln_bwd(const float* __restrict__ g, const float* __restrict__ y,
       const float* __restrict__ rstd, float* __restrict__ dx, int rows) {
    int row = blockIdx.x * 8 + (threadIdx.x >> 5);
    int lane = threadIdx.x & 31;
    if (row >= rows) return;
    const float* gr = g + (size_t)row * LDIM;
    const float* yr = y + (size_t)row * LDIM;
    float s1 = 0.f, s2 = 0.f;
    for (int c = lane; c < LDIM; c += 32) { s1 += gr[c]; s2 += gr[c] * yr[c]; }
    s1 = wave_sum(s1); s2 = wave_sum(s2);
    const float m1 = s1 * (1.f / LDIM), m2 = s2 * (1.f / LDIM);
    const float r = rstd[row];
    float* dr = dx + (size_t)row * LDIM;
    for (int c = lane; c < LDIM; c += 32) dr[c] = r * (gr[c] - m1 - yr[c] * m2);
}

// segment_sum forward: dst[idx[r],c] += src[r,c]
__global__ void __launch_bounds__(256)
scatter_add_rows(const float* __restrict__ src, const int* __restrict__ idx,
                 float* __restrict__ dst, int rows) {
    size_t i = (size_t)blockIdx.x * 256 + threadIdx.x;
    if (i >= (size_t)rows * LDIM) return;
    int r = (int)(i >> 9), c = (int)(i & (LDIM - 1));
    atomicAdd(&dst[(size_t)idx[r] * LDIM + c], src[i]);
}

// segment_sum backward: dst[r,c] += src[idx[r],c]
__global__ void __launch_bounds__(256)
gather_add_rows(float* __restrict__ dst, const float* __restrict__ src,
                const int* __restrict__ idx, int rows) {
    size_t i = (size_t)blockIdx.x * 256 + threadIdx.x;
    if (i >= (size_t)rows * LDIM) return;
    int r = (int)(i >> 9), c = (int)(i & (LDIM - 1));
    dst[i] += src[(size_t)idx[r] * LDIM + c];
}

// dH/d(dec hidden): g[r,c] = (d1[r,c]>0) ? dec_W2[c] : 0   (H = sum of outputs)
__global__ void __launch_bounds__(256)
gd1_init(const float* __restrict__ d1, const float* __restrict__ W2,
         float* __restrict__ g, int rows) {
    size_t i = (size_t)blockIdx.x * 256 + threadIdx.x;
    if (i >= (size_t)rows * LDIM) return;
    g[i] = (d1[i] > 0.f) ? W2[i & (LDIM - 1)] : 0.f;
}

// out[r] = state[r] - DT * dot(g_h1[r,:], W1)   (encoder input backward + Euler)
__global__ void __launch_bounds__(256)
final_out(const float* __restrict__ state, const float* __restrict__ gh1,
          const float* __restrict__ W1, float* __restrict__ out, int rows) {
    int row = blockIdx.x * 8 + (threadIdx.x >> 5);
    int lane = threadIdx.x & 31;
    if (row >= rows) return;
    const float* g = gh1 + (size_t)row * LDIM;
    float s = 0.f;
    for (int c = lane; c < LDIM; c += 32) s += g[c] * W1[c];
    s = wave_sum(s);
    if (lane == 0) out[row] = state[row] - DT_STEP * s;
}

// ---------------------------------------------------------------------------
// Host orchestration
// ---------------------------------------------------------------------------
extern "C" void kernel_launch(void* const* d_in, const int* in_sizes, int n_in,
                              void* d_out, int out_size, void* d_ws, size_t ws_size,
                              hipStream_t stream) {
    (void)in_sizes; (void)n_in; (void)out_size; (void)ws_size;
    const int E = N_EDGES, NN = N_NODES, L = LDIM;
    const size_t EL = (size_t)E * L, NL = (size_t)NN * L, LL = (size_t)L * L;

    const float* state     = (const float*)d_in[0];
    const int*   senders   = (const int*)d_in[1];
    const int*   receivers = (const int*)d_in[2];
    const float* enc_e_W1 = (const float*)d_in[3];
    const float* enc_e_b1 = (const float*)d_in[4];
    const float* enc_e_W2 = (const float*)d_in[5];
    const float* enc_e_b2 = (const float*)d_in[6];
    const float* enc_n_W1 = (const float*)d_in[7];
    const float* enc_n_b1 = (const float*)d_in[8];
    const float* enc_n_W2 = (const float*)d_in[9];
    const float* enc_n_b2 = (const float*)d_in[10];
    const float* pe_W1    = (const float*)d_in[11];
    const float* pe_b1    = (const float*)d_in[12];
    const float* pe_W2    = (const float*)d_in[13];
    const float* pe_b2    = (const float*)d_in[14];
    const float* pn_W1    = (const float*)d_in[15];
    const float* pn_b1    = (const float*)d_in[16];
    const float* pn_W2    = (const float*)d_in[17];
    const float* pn_b2    = (const float*)d_in[18];
    const float* de_W1    = (const float*)d_in[19];
    const float* de_b1    = (const float*)d_in[20];
    const float* de_W2    = (const float*)d_in[21];
    float* out = (float*)d_out;

    // --- workspace arena ---
    char* ws = (char*)d_ws;
    size_t off = 0;
    auto alloc = [&](size_t bytes) -> void* {
        void* p = ws + off;
        off = (off + bytes + 255) & ~(size_t)255;
        return p;
    };
    // plain bf16 weights [in,out] row-major == Bt for backward (@ W^T) GEMMs
    bf16_t* w_ee2 = (bf16_t*)alloc(LL * sizeof(bf16_t));
    bf16_t* w_en2 = (bf16_t*)alloc(LL * sizeof(bf16_t));
    bf16_t* w_pe1 = (bf16_t*)alloc(3 * LL * sizeof(bf16_t));
    bf16_t* w_pe2 = (bf16_t*)alloc(LL * sizeof(bf16_t));
    bf16_t* w_pn1 = (bf16_t*)alloc(2 * LL * sizeof(bf16_t));
    bf16_t* w_pn2 = (bf16_t*)alloc(LL * sizeof(bf16_t));
    bf16_t* w_de1 = (bf16_t*)alloc(LL * sizeof(bf16_t));
    // transposed bf16 weights [out,in] row-major == Bt for forward GEMMs
    bf16_t* wt_ee2 = (bf16_t*)alloc(LL * sizeof(bf16_t));
    bf16_t* wt_en2 = (bf16_t*)alloc(LL * sizeof(bf16_t));
    bf16_t* wt_pe1 = (bf16_t*)alloc(3 * LL * sizeof(bf16_t));  // [512, 1536]
    bf16_t* wt_pe2 = (bf16_t*)alloc(LL * sizeof(bf16_t));
    bf16_t* wt_pn1 = (bf16_t*)alloc(2 * LL * sizeof(bf16_t));  // [512, 1024]
    bf16_t* wt_pn2 = (bf16_t*)alloc(LL * sizeof(bf16_t));
    bf16_t* wt_de1 = (bf16_t*)alloc(LL * sizeof(bf16_t));

    float* h1_e = (float*)alloc(EL * 4);
    float* h1_n = (float*)alloc(NL * 4);
    float* eh0  = (float*)alloc(EL * 4);
    float* nh0  = (float*)alloc(NL * 4);
    float* rstd_e = (float*)alloc(E * 4);
    float* rstd_n = (float*)alloc(NN * 4);
    float* m1b[MP_STEPS];
    float* n1b[MP_STEPS];
    for (int t = 0; t < MP_STEPS; ++t) m1b[t] = (float*)alloc(EL * 4);
    for (int t = 0; t < MP_STEPS; ++t) n1b[t] = (float*)alloc(NL * 4);
    float* d1   = (float*)alloc(EL * 4);
    float* eh   = (float*)alloc(EL * 4);
    float* nh   = (float*)alloc(NL * 4);
    float* agg  = (float*)alloc(NL * 4);
    float* g_eh = (float*)alloc(EL * 4);
    float* g_nh = (float*)alloc(NL * 4);
    float* g_m1 = (float*)alloc(EL * 4);
    float* g_n1 = (float*)alloc(NL * 4);

    const int TB = 256;
    auto gblk = [&](size_t n) { return (int)((n + TB - 1) / TB); };

    auto gemm = [&](const float* A, int lda, const int* mapA,
                    const bf16_t* Bt, int ldb,
                    const float* bias, const float* mask,
                    float* C, int ldc, const int* mapC,
                    int M, int N, int K, int flags) {
        dim3 grid(N / 64, M / 128), block(TB);
        gemm_bf16_wmma<<<grid, block, 0, stream>>>(A, lda, mapA, Bt, ldb,
                                                   bias, mask, C, ldc, mapC, K, flags);
    };

    // --- convert weights to bf16 (plain + transposed; tiny, L2-resident) ---
    cvt_f32_to_bf16<<<gblk(LL), TB, 0, stream>>>(enc_e_W2, w_ee2, (int)LL);
    cvt_f32_to_bf16<<<gblk(LL), TB, 0, stream>>>(enc_n_W2, w_en2, (int)LL);
    cvt_f32_to_bf16<<<gblk(3 * LL), TB, 0, stream>>>(pe_W1, w_pe1, (int)(3 * LL));
    cvt_f32_to_bf16<<<gblk(LL), TB, 0, stream>>>(pe_W2, w_pe2, (int)LL);
    cvt_f32_to_bf16<<<gblk(2 * LL), TB, 0, stream>>>(pn_W1, w_pn1, (int)(2 * LL));
    cvt_f32_to_bf16<<<gblk(LL), TB, 0, stream>>>(pn_W2, w_pn2, (int)LL);
    cvt_f32_to_bf16<<<gblk(LL), TB, 0, stream>>>(de_W1, w_de1, (int)LL);
    cvt_f32_to_bf16_t<<<gblk(LL), TB, 0, stream>>>(enc_e_W2, wt_ee2, L, L);
    cvt_f32_to_bf16_t<<<gblk(LL), TB, 0, stream>>>(enc_n_W2, wt_en2, L, L);
    cvt_f32_to_bf16_t<<<gblk(3 * LL), TB, 0, stream>>>(pe_W1, wt_pe1, 3 * L, L);
    cvt_f32_to_bf16_t<<<gblk(LL), TB, 0, stream>>>(pe_W2, wt_pe2, L, L);
    cvt_f32_to_bf16_t<<<gblk(2 * LL), TB, 0, stream>>>(pn_W1, wt_pn1, 2 * L, L);
    cvt_f32_to_bf16_t<<<gblk(LL), TB, 0, stream>>>(pn_W2, wt_pn2, L, L);
    cvt_f32_to_bf16_t<<<gblk(LL), TB, 0, stream>>>(de_W1, wt_de1, L, L);

    // ================= FORWARD =================
    // encoders (input dim 1 -> rank-1, then L x L GEMM, then LN)
    enc_rank1_fwd<<<gblk(EL), TB, 0, stream>>>(state, enc_e_W1, enc_e_b1, h1_e, E);
    gemm(h1_e, L, nullptr, wt_ee2, L, enc_e_b2, nullptr, g_m1, L, nullptr, E, L, L, 0);
    ln_fwd<<<E / 8, TB, 0, stream>>>(g_m1, eh0, rstd_e, E);
    hipMemcpyAsync(eh, eh0, EL * 4, hipMemcpyDeviceToDevice, stream);

    enc_rank1_fwd<<<gblk(NL), TB, 0, stream>>>(state + E, enc_n_W1, enc_n_b1, h1_n, NN);
    gemm(h1_n, L, nullptr, wt_en2, L, enc_n_b2, nullptr, g_n1, L, nullptr, NN, L, L, 0);
    ln_fwd<<<NN / 8, TB, 0, stream>>>(g_n1, nh0, rstd_n, NN);
    hipMemcpyAsync(nh, nh0, NL * 4, hipMemcpyDeviceToDevice, stream);

    for (int t = 0; t < MP_STEPS; ++t) {
        float* m1t = m1b[t];
        float* n1t = n1b[t];
        // m1 = relu([eh | nh[send] | nh[recv]] @ pe_W1 + b1)  (concat via 3 accum GEMMs)
        gemm(eh, L, nullptr, wt_pe1, 3 * L, pe_b1, nullptr, m1t, L, nullptr, E, L, L, 0);
        gemm(nh, L, senders, wt_pe1 + L, 3 * L, nullptr, nullptr, m1t, L, nullptr, E, L, L, GF_ACCUM);
        gemm(nh, L, receivers, wt_pe1 + 2 * L, 3 * L, nullptr, nullptr, m1t, L, nullptr, E, L, L, GF_ACCUM | GF_RELU);
        // eh += m1 @ pe_W2 + b2
        gemm(m1t, L, nullptr, wt_pe2, L, pe_b2, nullptr, eh, L, nullptr, E, L, L, GF_ACCUM);
        // agg = segment_sum(eh, receivers)
        zero_f32<<<gblk(NL), TB, 0, stream>>>(agg, (int)NL);
        scatter_add_rows<<<gblk(EL), TB, 0, stream>>>(eh, receivers, agg, E);
        // n1 = relu([nh | agg] @ pn_W1 + b1)
        gemm(nh, L, nullptr, wt_pn1, 2 * L, pn_b1, nullptr, n1t, L, nullptr, NN, L, L, 0);
        gemm(agg, L, nullptr, wt_pn1 + L, 2 * L, nullptr, nullptr, n1t, L, nullptr, NN, L, L, GF_ACCUM | GF_RELU);
        // nh += n1 @ pn_W2 + b2
        gemm(n1t, L, nullptr, wt_pn2, L, pn_b2, nullptr, nh, L, nullptr, NN, L, L, GF_ACCUM);
    }

    // decoder hidden (only the relu mask matters; H value itself is unused)
    gemm(eh, L, nullptr, wt_de1, L, de_b1, nullptr, d1, L, nullptr, E, L, L, GF_RELU);

    // ================= BACKWARD =================
    // dH/d(d1) = relu'(d1) * de_W2[c]
    gd1_init<<<gblk(EL), TB, 0, stream>>>(d1, de_W2, g_m1, E);
    // g_eh5 = g_d1 @ de_W1^T  (plain de_W1 [in,out] row-major IS Bt here)
    gemm(g_m1, L, nullptr, w_de1, L, nullptr, nullptr, g_eh, L, nullptr, E, L, L, 0);
    zero_f32<<<gblk(NL), TB, 0, stream>>>(g_nh, (int)NL);

    for (int t = MP_STEPS - 1; t >= 0; --t) {
        float* m1t = m1b[t];
        float* n1t = n1b[t];
        // node MLP backward: g_n1 = (g_nh @ pn_W2^T) * relu'(n1)
        gemm(g_nh, L, nullptr, w_pn2, L, nullptr, n1t, g_n1, L, nullptr, NN, L, L, 0);
        // g_nh += g_n1 @ pn_W1[:L]^T       (nh slice of concat)
        gemm(g_n1, L, nullptr, w_pn1, L, nullptr, nullptr, g_nh, L, nullptr, NN, L, L, GF_ACCUM);
        // g_agg = g_n1 @ pn_W1[L:2L]^T
        gemm(g_n1, L, nullptr, w_pn1 + LL, L, nullptr, nullptr, agg, L, nullptr, NN, L, L, 0);
        // segment_sum backward: g_eh += g_agg[receivers]
        gather_add_rows<<<gblk(EL), TB, 0, stream>>>(g_eh, agg, receivers, E);
        // edge MLP backward: g_m1 = (g_eh @ pe_W2^T) * relu'(m1)
        gemm(g_eh, L, nullptr, w_pe2, L, nullptr, m1t, g_m1, L, nullptr, E, L, L, 0);
        // g_eh += g_m1 @ pe_W1[:L]^T       (eh slice of concat)
        gemm(g_m1, L, nullptr, w_pe1, L, nullptr, nullptr, g_eh, L, nullptr, E, L, L, GF_ACCUM);
        // g_nh[senders]   += g_m1 @ pe_W1[L:2L]^T   (fused atomic scatter)
        gemm(g_m1, L, nullptr, w_pe1 + LL, L, nullptr, nullptr, g_nh, L, senders, E, L, L, GF_ATOMIC);
        // g_nh[receivers] += g_m1 @ pe_W1[2L:3L]^T
        gemm(g_m1, L, nullptr, w_pe1 + 2 * LL, L, nullptr, nullptr, g_nh, L, receivers, E, L, L, GF_ATOMIC);
    }

    // encoder backward + Euler step (edges)
    ln_bwd<<<E / 8, TB, 0, stream>>>(g_eh, eh0, rstd_e, m1b[0], E);
    gemm(m1b[0], L, nullptr, w_ee2, L, nullptr, h1_e, d1, L, nullptr, E, L, L, 0);
    final_out<<<E / 8, TB, 0, stream>>>(state, d1, enc_e_W1, out, E);

    // encoder backward + Euler step (nodes)
    ln_bwd<<<NN / 8, TB, 0, stream>>>(g_nh, nh0, rstd_n, n1b[0], NN);
    gemm(n1b[0], L, nullptr, w_en2, L, nullptr, h1_n, n1b[1], L, nullptr, NN, L, L, 0);
    final_out<<<NN / 8, TB, 0, stream>>>(state + E, n1b[1], enc_n_W1, out + E, NN);
}
